// AmpereFP8Linear_16544214024535
// MI455X (gfx1250) — matrix operand order repaired and checked by reading, not compiled, not run on hardware
//
#include <hip/hip_runtime.h>
#include <stdint.h>

// ---------------------------------------------------------------------------
// out[m,n] = (sum_k act[m,k] * (fp16)weight[k,n]) * scale[n],  fp16 result
// M=32, K=8192, N=28672. HBM-bound on the 235MB int8 weight stream (~10us).
// Core op: v_wmma_f32_16x16x32_f16 (wave32); weight staged into LDS with
// GLOBAL_LOAD_ASYNC_TO_LDS_B128 (ASYNCcnt); hand-rolled split barrier so
// in-flight A loads (LOADcnt) are never drained at the workgroup barrier.
// ---------------------------------------------------------------------------

typedef _Float16 v16h __attribute__((ext_vector_type(16)));
typedef _Float16 v8h  __attribute__((ext_vector_type(8)));
typedef float    v8f  __attribute__((ext_vector_type(8)));
typedef int      v4i  __attribute__((ext_vector_type(4)));

#define GEMM_M 32
#define GEMM_K 8192
#define GEMM_N 28672

#define BLOCK    256                 // 8 wave32s
#define COLS_WG  128                 // 8 waves x 16 columns
#define KSTEP    32                  // WMMA K per step
#define KTILES   (GEMM_K / KSTEP)    // 256 (even: loop unrolled by 2)
#define NSTAGE   4                   // LDS ring depth (3 tiles in flight)

// LDS tile layout for the 32x128 int8 weight tile:
//   addr(r, c) = (r & 15) * 144 + (r >> 4) * 2336 + c
// Row stride 144 keeps b128 async writes 16B-aligned; the +32B skew between
// the two 16-row halves (2336 = 16*144 + 32) applies to both lane halves alike:
// in each gather load lanes 16-31 read 8 rows (1152B) past lanes 0-15.
#define LDSROW     144
#define HALFSKEW   2336
#define STAGEBYTES 4640              // > 2336 + 15*144 + 128, 16B aligned

__device__ __forceinline__ int lds_ofs(int r, int c) {
    return (r & 15) * LDSROW + (r >> 4) * HALFSKEW + c;
}

// ---- CDNA5 async copy: global -> LDS, 16B per lane, tracked by ASYNCcnt ----
__device__ __forceinline__ void async_copy_b128(uint32_t lds_byte_addr,
                                                const int8_t* gsrc) {
    asm volatile("global_load_async_to_lds_b128 %0, %1, off th:TH_LOAD_NT"
                 :: "v"(lds_byte_addr), "v"((uint64_t)(uintptr_t)gsrc)
                 : "memory");
}

// Workgroup barrier for the staging pipeline. Waits only what the LDS ring
// actually needs: DScnt==0 (my gathers of the slot being recycled retired,
// already drained by the cvt consumers) and ASYNCcnt<=2 (my oldest in-flight
// async tile has landed). Deliberately does NOT touch LOADcnt, so the
// software-pipelined A loads stay in flight across the barrier.
__device__ __forceinline__ void pipeline_barrier() {
    asm volatile("s_wait_dscnt 0x0\n\t"
                 "s_wait_asynccnt 0x2\n\t"
                 "s_barrier_signal -1\n\t"
                 "s_barrier_wait -1"
                 ::: "memory");
}

__global__ __launch_bounds__(BLOCK, 2)
void int8w_f16a_gemm(const _Float16* __restrict__ act,
                     const int8_t*  __restrict__ wgt,
                     const _Float16* __restrict__ scale,
                     _Float16* __restrict__ out)
{
    __shared__ int8_t lw[NSTAGE * STAGEBYTES];

    const int tid  = threadIdx.x;
    const int lane = tid & 31;
    const int wave = tid >> 5;
    const int n0   = blockIdx.x * COLS_WG;

    // ---- staging map: each thread moves one 16B chunk of the 32x128 tile.
    // 8 consecutive threads cover one 128B row -> fully coalesced 128B rows.
    const int srow = tid >> 3;              // 0..31 (K row within tile)
    const int scol = (tid & 7) << 4;        // 0..112 (byte column)
    const int8_t* gstage = wgt + (size_t)srow * GEMM_N + (size_t)(n0 + scol);
    const uint32_t lstage =
        (uint32_t)(uintptr_t)&lw[lds_ofs(srow, scol)];   // LDS byte address

    // ---- WMMA fragment coordinates (ISA 7.12.2 layouts)
    const int half16 = lane >> 4;           // lane group 0/1
    const int nloc   = lane & 15;
    const int colL   = (wave << 4) + nloc;  // column inside WG tile
    const int colG   = n0 + colL;           // global column
    const int aOff   = half16 << 3;         // A: K sub-offset {0,8}

    v8f c0 = {};                            // M rows 0..15
    v8f c1 = {};                            // M rows 16..31

    // stage weight tile kt into ring slot kt&3 (one async b128 per thread)
    auto stage = [&](int kt) {
        const int buf = kt & (NSTAGE - 1);
        const int8_t* src = gstage + (size_t)kt * KSTEP * GEMM_N;
        async_copy_b128(lstage + buf * STAGEBYTES, src);
    };

    // A fragment (16x32 fp16): lane m=nloc holds K {aOff..+7, 16+aOff..+7}
    auto loadA = [&](int kt, v16h& a0, v16h& a1) {
        const _Float16* p0 = act + (size_t)nloc * GEMM_K + kt * KSTEP + aOff;
        v8h lo0 = *(const v8h*)(p0);
        v8h hi0 = *(const v8h*)(p0 + 16);
        const _Float16* p1 = p0 + (size_t)16 * GEMM_K;         // M rows 16..31
        v8h lo1 = *(const v8h*)(p1);
        v8h hi1 = *(const v8h*)(p1 + 16);
        a0 = __builtin_shufflevector(lo0, hi0, 0,1,2,3,4,5,6,7,8,9,10,11,12,13,14,15);
        a1 = __builtin_shufflevector(lo1, hi1, 0,1,2,3,4,5,6,7,8,9,10,11,12,13,14,15);
    };

    // One pipeline step: consume A regs (ac*), prefetch next A into (an*).
    auto step = [&](int kt, v16h& ac0, v16h& ac1, v16h& an0, v16h& an1) {
        if (kt + 1 < KTILES) loadA(kt + 1, an0, an1);   // uniform branch

        pipeline_barrier();          // slot kt ready everywhere; slot (kt-1)
                                     // free for rewrite; A loads still in flight

        if (kt + NSTAGE - 1 < KTILES) stage(kt + NSTAGE - 1);  // writes (kt-1)&3

        // B fragment (32x16 fp16): lane holds K {half16*8..+7, 16+half16*8..+7} of col colL.
        // Column gather from LDS tile, immediate offsets, dequant int8 -> fp16.
        const int buf = kt & (NSTAGE - 1);
        const int8_t* lp = lw + buf * STAGEBYTES + half16 * 8 * LDSROW + colL;
        v16h b;
        #pragma unroll
        for (int j = 0; j < 16; ++j)
            b[j] = (_Float16)lp[j < 8 ? j * LDSROW : HALFSKEW + (j - 8) * LDSROW];            // ds_load_i8 + cvt

        c0 = __builtin_amdgcn_wmma_f32_16x16x32_f16(false, ac0, false, b,
                                                    (short)0, c0, false, false);
        c1 = __builtin_amdgcn_wmma_f32_16x16x32_f16(false, ac1, false, b,
                                                    (short)0, c1, false, false);
    };

    // fill the async pipeline: tiles 0,1,2 in flight
    #pragma unroll
    for (int p = 0; p < NSTAGE - 1; ++p) stage(p);

    v16h aA0, aA1, aB0, aB1;
    loadA(0, aA0, aA1);

    // 2x unrolled with A-register ping-pong: no per-iteration register copies
    for (int kt = 0; kt < KTILES; kt += 2) {
        step(kt,     aA0, aA1, aB0, aB1);
        step(kt + 1, aB0, aB1, aA0, aA1);
    }

    // ---- epilogue: per-column scale, fp16 store
    // C/D layout: element r of v8f -> M = r + 8*half16 (+16 for c1), N = nloc.
    const float s = (float)scale[colG];
    #pragma unroll
    for (int r = 0; r < 8; ++r) {
        const int m0 = r + (half16 << 3);
        __builtin_nontemporal_store((_Float16)(c0[r] * s),
                                    out + (size_t)m0 * GEMM_N + colG);
        __builtin_nontemporal_store((_Float16)(c1[r] * s),
                                    out + (size_t)(m0 + 16) * GEMM_N + colG);
    }
}

extern "C" void kernel_launch(void* const* d_in, const int* in_sizes, int n_in,
                              void* d_out, int out_size, void* d_ws, size_t ws_size,
                              hipStream_t stream) {
    const _Float16* act   = (const _Float16*)d_in[0];
    const int8_t*   wgt   = (const int8_t*)d_in[1];    // int8 weight, K x N
    const _Float16* scale = (const _Float16*)d_in[2];
    _Float16*       out   = (_Float16*)d_out;

    (void)in_sizes; (void)n_in; (void)out_size; (void)d_ws; (void)ws_size;

    dim3 grid(GEMM_N / COLS_WG);   // 224 workgroups
    int8w_f16a_gemm<<<grid, BLOCK, 0, stream>>>(act, wgt, scale, out);
}
